// MultiHeadAttention_42417097017079
// MI455X (gfx1250) — matrix operand order repaired and checked by reading, not compiled, not run on hardware
//
#include <hip/hip_runtime.h>
#include <hip/hip_bf16.h>

// Problem constants (from reference)
#define NHEAD 16
#define DHEAD 64
#define DMODEL 1024
#define SEQ 2048
#define BATCH 2
#define NTOK (SEQ * BATCH) // 4096

typedef __bf16 bf16_t;
typedef __bf16 v16bf __attribute__((ext_vector_type(16)));
typedef __bf16 v8bf  __attribute__((ext_vector_type(8)));
typedef float  v8f   __attribute__((ext_vector_type(8)));

union V16U { v16bf v; v8bf h[2]; };

// ---------------------------------------------------------------------------
// Optional CDNA5 async global->LDS path (ASYNCcnt-tracked), with safe fallback
// Signature leaked by clang diagnostic: param0 = int __vector_size(16) ptr in
// global (__device__/as1) address space; 4 args total.
// ---------------------------------------------------------------------------
#if defined(__has_builtin)
#if __has_builtin(__builtin_amdgcn_global_load_async_to_lds_b128)
#define USE_ASYNC_LDS 1
#endif
#if __has_builtin(__builtin_amdgcn_s_wait_asynccnt)
#define HAVE_WAIT_ASYNC_BUILTIN 1
#endif
#endif

typedef int v4i_vs __attribute__((vector_size(4 * sizeof(int))));
typedef __attribute__((address_space(1))) v4i_vs gas_v4i;
typedef __attribute__((address_space(3))) v4i_vs las_v4i;

__device__ __forceinline__ void cp16_g2s(bf16_t* lds_dst, const bf16_t* gsrc) {
#if defined(USE_ASYNC_LDS)
    __builtin_amdgcn_global_load_async_to_lds_b128(
        (gas_v4i*)gsrc, (las_v4i*)lds_dst, /*imm offset*/ 0, /*cpol*/ 0);
#else
    *(uint4*)lds_dst = *(const uint4*)gsrc;
#endif
}

__device__ __forceinline__ void wait_async_all() {
#if defined(USE_ASYNC_LDS)
#if defined(HAVE_WAIT_ASYNC_BUILTIN)
    __builtin_amdgcn_s_wait_asynccnt(0);
#else
    asm volatile("s_wait_asynccnt 0" ::: "memory");
#endif
#endif
}

// ---------------------------------------------------------------------------
// WMMA helpers (CDNA5 wave32, 16x16x32 bf16 -> f32)
// ---------------------------------------------------------------------------
__device__ __forceinline__ v8f wmma_bf16(v16bf a, v16bf b, v8f c) {
    return __builtin_amdgcn_wmma_f32_16x16x32_bf16(
        /*neg_a=*/false, a, /*neg_b=*/false, b,
        /*c_mod=*/(short)0, c, /*reuse_a=*/false, /*reuse_b=*/false);
}

// A-matrix fragment (16x32, row-major source with leading dim ld):
//   lanes 0-15 : row = row0+lane,  K = k0+0..7 (h[0]),  k0+16..23 (h[1])
//   lanes 16-31: row = row0+lane-16, K = k0+8..15 (h[0]), k0+24..31 (h[1])
template <typename PT>
__device__ __forceinline__ v16bf load_a_frag(const PT* base, int ld, int row0,
                                             int k0, int lane) {
    int m = lane & 15, hf = lane >> 4;
    const PT* p = base + (size_t)(row0 + m) * ld + k0 + hf * 8;
    V16U u;
    u.h[0] = *(const v8bf*)(p);
    u.h[1] = *(const v8bf*)(p + 16);
    return u.v;
}

// B-matrix fragment for C = A * (Bsrc)^T, Bsrc row-major [N][K] with ld:
//   B[k][n] = Bsrc[n][k]
//   lanes 0-15 : n = n0+lane,  K = k0+0..7 (h[0]),  k0+16..23 (h[1])
//   lanes 16-31: n = n0+lane-16, K = k0+8..15 (h[0]), k0+24..31 (h[1])
template <typename PT>
__device__ __forceinline__ v16bf load_bt_frag(const PT* base, int ld, int n0,
                                              int k0, int lane) {
    int n = lane & 15, hf = lane >> 4;
    const PT* p = base + (size_t)(n0 + n) * ld + k0 + hf * 8;
    V16U u;
    u.h[0] = *(const v8bf*)(p);
    u.h[1] = *(const v8bf*)(p + 16);
    return u.v;
}

// ---------------------------------------------------------------------------
// fp32 -> bf16 conversion
// ---------------------------------------------------------------------------
__global__ void cvt_f32_bf16(const float* __restrict__ in,
                             bf16_t* __restrict__ out, int n) {
    int i = blockIdx.x * blockDim.x + threadIdx.x;
    if (i < n) out[i] = (bf16_t)in[i];
}

// ---------------------------------------------------------------------------
// GEMM: C[M=NTOK, N=DMODEL] = A[M,K] * Bm[N,K]^T  (bf16 in, f32 acc)
// MODE 0: store bf16 to [B,H,S,D]      (Q / K projection)
// MODE 1: store bf16 to [B,H,D,S]      (V projection, transposed)
// MODE 2: store f32  to row-major [M,N] (final output projection)
// Block: 256 threads = 8 waves, tile 128(M) x 64(N); wave tile 32x32.
// ---------------------------------------------------------------------------
template <int MODE>
__global__ __launch_bounds__(256) void gemm_bt(const bf16_t* __restrict__ A,
                                               const bf16_t* __restrict__ Bm,
                                               bf16_t* __restrict__ Cb,
                                               float* __restrict__ Cf,
                                               int Kdim) {
    int lane = threadIdx.x & 31;
    int w = threadIdx.x >> 5; // 0..7
    int m0 = blockIdx.y * 128 + (w & 3) * 32;
    int n0 = blockIdx.x * 64 + (w >> 2) * 32;

    v8f c00 = {}, c01 = {}, c10 = {}, c11 = {};
    for (int k0 = 0; k0 < Kdim; k0 += 32) {
        if (k0 + 32 < Kdim) // hint the next A rows into cache
            __builtin_prefetch(A + (size_t)(m0 + (lane & 15)) * Kdim + k0 + 32, 0, 1);
        v16bf a0 = load_a_frag(A, Kdim, m0, k0, lane);
        v16bf a1 = load_a_frag(A, Kdim, m0 + 16, k0, lane);
        v16bf b0 = load_bt_frag(Bm, Kdim, n0, k0, lane);
        v16bf b1 = load_bt_frag(Bm, Kdim, n0 + 16, k0, lane);
        c00 = wmma_bf16(a0, b0, c00);
        c01 = wmma_bf16(a0, b1, c01);
        c10 = wmma_bf16(a1, b0, c10);
        c11 = wmma_bf16(a1, b1, c11);
    }

    int col = lane & 15, hf = lane >> 4;
    v8f cc[2][2] = {{c00, c01}, {c10, c11}};
#pragma unroll
    for (int mt = 0; mt < 2; ++mt)
#pragma unroll
        for (int nt = 0; nt < 2; ++nt)
#pragma unroll
            for (int r = 0; r < 8; ++r) {
                int row = m0 + mt * 16 + hf * 8 + r; // token index t = s*BS+b
                int cn = n0 + nt * 16 + col;         // feature j = h*64+d
                float val = cc[mt][nt][r];
                if (MODE == 2) {
                    Cf[(size_t)row * DMODEL + cn] = val;
                } else {
                    int s = row >> 1, b = row & 1; // BATCH == 2
                    int hh = cn >> 6, d = cn & 63; // DHEAD == 64
                    if (MODE == 0)
                        Cb[(((size_t)(b * NHEAD + hh)) * SEQ + s) * DHEAD + d] =
                            (bf16_t)val;
                    else
                        Cb[(((size_t)(b * NHEAD + hh)) * DHEAD + d) * SEQ + s] =
                            (bf16_t)val;
                }
            }
}

// ---------------------------------------------------------------------------
// Flash attention (no score scaling, per reference).
// Grid: (SEQ/64, BATCH*NHEAD). Block: 128 threads = 4 waves.
// Wave: 16 query rows x full 64-dim output; loops 64-key blocks with
// online softmax. Qh,Kh: [B,H,S,D] bf16. Vt: [B,H,D,S] bf16.
// K/V tiles are staged in LDS (double-buffered, async when available).
// Output Attn: row-major [t = s*BS+b][h*64+d] bf16.
// ---------------------------------------------------------------------------
__device__ __forceinline__ void stage_kv(bf16_t* Kl, bf16_t* Vl,
                                         const bf16_t* Kp, const bf16_t* Vp,
                                         int kb, int tid) {
    // K tile: rows kb..kb+63 of [S][64]; V tile: 64 d-rows, cols kb..kb+63
    // 64x64 bf16 = 512 chunks of 16B each; 128 threads x 4 chunks.
#pragma unroll
    for (int i = 0; i < 4; ++i) {
        int c = tid + i * 128;          // 0..511
        int row = c >> 3;               // 0..63
        int off = (c & 7) * 8;          // 0..56 elements
        cp16_g2s(Kl + row * 64 + off, Kp + (size_t)(kb + row) * DHEAD + off);
        cp16_g2s(Vl + row * 64 + off, Vp + (size_t)row * SEQ + kb + off);
    }
}

__global__ __launch_bounds__(128) void flash_attn(const bf16_t* __restrict__ Qh,
                                                  const bf16_t* __restrict__ Kh,
                                                  const bf16_t* __restrict__ Vt,
                                                  bf16_t* __restrict__ Attn) {
    int tid = threadIdx.x;
    int lane = tid & 31;
    int wv = tid >> 5;   // 0..3
    int bh = blockIdx.y; // b*NHEAD + h
    int q0 = blockIdx.x * 64 + wv * 16;

    const bf16_t* Qp = Qh + (size_t)bh * SEQ * DHEAD;
    const bf16_t* Kp = Kh + (size_t)bh * SEQ * DHEAD;
    const bf16_t* Vp = Vt + (size_t)bh * DHEAD * SEQ;

    __shared__ __align__(16) bf16_t Kl[2][64 * 64]; // [key][d]
    __shared__ __align__(16) bf16_t Vl[2][64 * 64]; // [d][key]
    __shared__ __align__(16) bf16_t Pl[4][16 * 64]; // wave-private P tiles

    v8f o0 = {}, o1 = {}, o2 = {}, o3 = {};
    float mrow[8], lrow[8];
#pragma unroll
    for (int r = 0; r < 8; ++r) { mrow[r] = -1e30f; lrow[r] = 0.0f; }

    // Q fragments for both d k-steps are loop-invariant
    v16bf qa0 = load_a_frag(Qp, DHEAD, q0, 0, lane);
    v16bf qa1 = load_a_frag(Qp, DHEAD, q0, 32, lane);

    // prologue: stage first K/V tile
    stage_kv(Kl[0], Vl[0], Kp, Vp, 0, tid);
    wait_async_all();
    __syncthreads();

    int buf = 0;
    for (int kb = 0; kb < SEQ; kb += 64) {
        // issue async staging of the NEXT tile into the other buffer
        if (kb + 64 < SEQ)
            stage_kv(Kl[buf ^ 1], Vl[buf ^ 1], Kp, Vp, kb + 64, tid);

        const bf16_t* kt_ = Kl[buf];
        const bf16_t* vt_ = Vl[buf];

        // ---- S = Q * K^T for this 16x64 tile (contraction over d=64) ----
        v8f s0 = {}, s1 = {}, s2 = {}, s3 = {};
#pragma unroll
        for (int kd = 0; kd < 2; ++kd) {
            v16bf qa = kd ? qa1 : qa0;
            v16bf b0 = load_bt_frag(kt_, 64, 0, kd * 32, lane);
            v16bf b1 = load_bt_frag(kt_, 64, 16, kd * 32, lane);
            v16bf b2 = load_bt_frag(kt_, 64, 32, kd * 32, lane);
            v16bf b3 = load_bt_frag(kt_, 64, 48, kd * 32, lane);
            s0 = wmma_bf16(qa, b0, s0);
            s1 = wmma_bf16(qa, b1, s1);
            s2 = wmma_bf16(qa, b2, s2);
            s3 = wmma_bf16(qa, b3, s3);
        }

        // ---- online softmax: row stats live per VGPR-slot per half-wave ----
#pragma unroll
        for (int r = 0; r < 8; ++r) {
            float mx = fmaxf(fmaxf(s0[r], s1[r]), fmaxf(s2[r], s3[r]));
#pragma unroll
            for (int msk = 8; msk >= 1; msk >>= 1)
                mx = fmaxf(mx, __shfl_xor(mx, msk, 32));
            float mnew = fmaxf(mrow[r], mx);
            float scale = __expf(mrow[r] - mnew);
            mrow[r] = mnew;
            o0[r] *= scale; o1[r] *= scale; o2[r] *= scale; o3[r] *= scale;
            float p0 = __expf(s0[r] - mnew);
            float p1 = __expf(s1[r] - mnew);
            float p2 = __expf(s2[r] - mnew);
            float p3 = __expf(s3[r] - mnew);
            s0[r] = p0; s1[r] = p1; s2[r] = p2; s3[r] = p3;
            float sum = p0 + p1 + p2 + p3;
#pragma unroll
            for (int msk = 8; msk >= 1; msk >>= 1)
                sum += __shfl_xor(sum, msk, 32);
            lrow[r] = lrow[r] * scale + sum;
        }

        // ---- C-layout -> A-layout via wave-private LDS (bf16) ----
        int col = lane & 15, hf = lane >> 4;
        bf16_t* pl = Pl[wv];
#pragma unroll
        for (int r = 0; r < 8; ++r) {
            int row = hf * 8 + r;
            pl[row * 64 + 0 * 16 + col] = (bf16_t)s0[r];
            pl[row * 64 + 1 * 16 + col] = (bf16_t)s1[r];
            pl[row * 64 + 2 * 16 + col] = (bf16_t)s2[r];
            pl[row * 64 + 3 * 16 + col] = (bf16_t)s3[r];
        }
        __syncthreads(); // P visible (same-wave RAW; barrier also keeps waves in phase)

        // ---- O += P * V (contraction over the 64 keys of this block) ----
#pragma unroll
        for (int kt = 0; kt < 2; ++kt) {
            v16bf pa = load_a_frag(pl, 64, 0, kt * 32, lane);
            v16bf vb0 = load_bt_frag(vt_, 64, 0, kt * 32, lane);
            v16bf vb1 = load_bt_frag(vt_, 64, 16, kt * 32, lane);
            v16bf vb2 = load_bt_frag(vt_, 64, 32, kt * 32, lane);
            v16bf vb3 = load_bt_frag(vt_, 64, 48, kt * 32, lane);
            o0 = wmma_bf16(pa, vb0, o0);
            o1 = wmma_bf16(pa, vb1, o1);
            o2 = wmma_bf16(pa, vb2, o2);
            o3 = wmma_bf16(pa, vb3, o3);
        }

        // next tile's async copies must have landed before we flip buffers
        wait_async_all();
        __syncthreads();
        buf ^= 1;
    }

    // ---- normalize by row sums and store to Attn [t][h*64+d] ----
    int col = lane & 15, hf = lane >> 4;
    int b = bh >> 4, hh = bh & 15;
#pragma unroll
    for (int r = 0; r < 8; ++r) {
        float inv = 1.0f / lrow[r];
        int qrow = q0 + hf * 8 + r;
        size_t t = (size_t)qrow * BATCH + b;
        bf16_t* dst = Attn + t * DMODEL + (size_t)hh * DHEAD;
        dst[0 * 16 + col] = (bf16_t)(o0[r] * inv);
        dst[1 * 16 + col] = (bf16_t)(o1[r] * inv);
        dst[2 * 16 + col] = (bf16_t)(o2[r] * inv);
        dst[3 * 16 + col] = (bf16_t)(o3[r] * inv);
    }
}

// ---------------------------------------------------------------------------
// Host launch
// ---------------------------------------------------------------------------
extern "C" void kernel_launch(void* const* d_in, const int* in_sizes, int n_in,
                              void* d_out, int out_size, void* d_ws,
                              size_t ws_size, hipStream_t stream) {
    const float* q = (const float*)d_in[0];
    const float* k = (const float*)d_in[1];
    const float* v = (const float*)d_in[2];
    const float* Wq = (const float*)d_in[3];
    const float* Wk = (const float*)d_in[4];
    const float* Wv = (const float*)d_in[5];
    const float* Wo = (const float*)d_in[6];
    float* out = (float*)d_out;

    char* ws = (char*)d_ws;
    size_t off = 0;
    auto carve = [&](size_t bytes) -> bf16_t* {
        bf16_t* p = (bf16_t*)(ws + off);
        off += (bytes + 255) & ~(size_t)255;
        return p;
    };
    const size_t actB = (size_t)NTOK * DMODEL * sizeof(bf16_t); // 8 MB
    const size_t wB = (size_t)DMODEL * DMODEL * sizeof(bf16_t); // 2 MB
    bf16_t* Xq = carve(actB);
    bf16_t* Xk = carve(actB);
    bf16_t* Xv = carve(actB);
    bf16_t* Wqb = carve(wB);
    bf16_t* Wkb = carve(wB);
    bf16_t* Wvb = carve(wB);
    bf16_t* Wob = carve(wB);
    bf16_t* Qh = carve(actB); // [B,H,S,D]
    bf16_t* Kh = carve(actB); // [B,H,S,D]
    bf16_t* Vt = carve(actB); // [B,H,D,S]
    bf16_t* At = carve(actB); // [t][h*64+d]

    const int nAct = NTOK * DMODEL; // 4,194,304
    const int nW = DMODEL * DMODEL; // 1,048,576
    dim3 cvtB(256);
    cvt_f32_bf16<<<nAct / 256, cvtB, 0, stream>>>(q, Xq, nAct);
    cvt_f32_bf16<<<nAct / 256, cvtB, 0, stream>>>(k, Xk, nAct);
    cvt_f32_bf16<<<nAct / 256, cvtB, 0, stream>>>(v, Xv, nAct);
    cvt_f32_bf16<<<nW / 256, cvtB, 0, stream>>>(Wq, Wqb, nW);
    cvt_f32_bf16<<<nW / 256, cvtB, 0, stream>>>(Wk, Wkb, nW);
    cvt_f32_bf16<<<nW / 256, cvtB, 0, stream>>>(Wv, Wvb, nW);
    cvt_f32_bf16<<<nW / 256, cvtB, 0, stream>>>(Wo, Wob, nW);

    dim3 gGrid(DMODEL / 64, NTOK / 128); // (16, 32)
    dim3 gBlk(256);
    gemm_bt<0><<<gGrid, gBlk, 0, stream>>>(Xq, Wqb, Qh, nullptr, DMODEL);
    gemm_bt<0><<<gGrid, gBlk, 0, stream>>>(Xk, Wkb, Kh, nullptr, DMODEL);
    gemm_bt<1><<<gGrid, gBlk, 0, stream>>>(Xv, Wvb, Vt, nullptr, DMODEL);

    dim3 aGrid(SEQ / 64, BATCH * NHEAD); // (32, 32)
    dim3 aBlk(128);
    flash_attn<<<aGrid, aBlk, 0, stream>>>(Qh, Kh, Vt, At);

    gemm_bt<2><<<gGrid, gBlk, 0, stream>>>(At, Wob, nullptr, out, DMODEL);
}
